// CausalSelfAttention_58849641889799
// MI455X (gfx1250) — matrix operand order repaired and checked by reading, not compiled, not run on hardware
//
#include <hip/hip_runtime.h>

typedef unsigned short u16;
typedef __attribute__((ext_vector_type(16))) __bf16       v16bf;
typedef __attribute__((ext_vector_type(8)))  float        v8f;
typedef __attribute__((ext_vector_type(4)))  unsigned int v4u;
typedef __attribute__((ext_vector_type(4)))  unsigned int v4ui;
typedef __attribute__((ext_vector_type(8)))  int          v8i;
typedef __attribute__((ext_vector_type(4)))  int          v4i;

union Frag { v16bf bf; v4u u[2]; };

#define S_  2048
#define D_  1024
#define H_  16
#define HD_ 64
#define M_  4096   // B*S

__device__ __forceinline__ u16 f2bf(float f) {
  unsigned int u = __float_as_uint(f);
  u += 0x7FFFu + ((u >> 16) & 1u);          // round-to-nearest-even
  return (u16)(u >> 16);
}

__device__ __forceinline__ v8f wmma_bf16(const Frag& a, const Frag& b, v8f c) {
  return __builtin_amdgcn_wmma_f32_16x16x32_bf16(
      /*neg_a=*/false, a.bf, /*neg_b=*/false, b.bf,
      /*c_mod=*/(short)0, c, /*reuse_a=*/false, /*reuse_b=*/false);
}

// ---- Tensor Data Mover: DMA a (64 x 32) bf16 tile of a row-major
// 1024-strided matrix into LDS, packed row-major 32 elems (64B) per row.
// D# layout per CDNA5 ISA 8.3/8.4. g1 is loop-invariant; ga/lds vary.
__device__ __forceinline__ void tdm_issue(unsigned long long ga, unsigned lds_off,
                                          const v8i& g1) {
  v4ui g0;
  g0[0] = 1u;                                        // count=1, user descriptor
  g0[1] = lds_off;                                   // lds_addr (bytes)
  g0[2] = (unsigned)(ga & 0xFFFFFFFFu);              // global_addr[31:0]
  g0[3] = (unsigned)((ga >> 32) & 0x1FFFFFFu) | (2u << 30);  // ga[56:32] | type=2
  v4i z4 = {0, 0, 0, 0};
#if defined(__clang_major__) && __clang_major__ >= 23
  v8i z8 = {0, 0, 0, 0, 0, 0, 0, 0};
  __builtin_amdgcn_tensor_load_to_lds(g0, g1, z4, z4, z8, 0);
#else
  __builtin_amdgcn_tensor_load_to_lds(g0, g1, z4, z4, 0);
#endif
}

__device__ __forceinline__ v8i tdm_desc_g1() {
  v8i g1;
  g1[0] = 0x00010000;                                // data_size=2B; no mask/pad
  g1[1] = (int)(D_ << 16);                           // tensor_dim0 = 1024 (lo16)
  g1[2] = (int)(D_ << 16);                           // td0 hi=0 | tensor_dim1 lo16
  g1[3] = (int)(32 << 16);                           // td1 hi=0 | tile_dim0 = 32
  g1[4] = 64;                                        // tile_dim1 = 64 | tile_dim2 = 0
  g1[5] = D_;                                        // tensor_dim0_stride = 1024
  g1[6] = 0;
  g1[7] = 0;
  return g1;
}

// ---------------- stage 0: fp32 -> bf16 ----------------
__global__ void cvt_f32_bf16(const float* __restrict__ src, u16* __restrict__ dst, int n) {
  int i = blockIdx.x * blockDim.x + threadIdx.x;
  int stride = gridDim.x * blockDim.x;
  for (; i < n; i += stride) dst[i] = f2bf(src[i]);
}

// ============= shared GEMM body: C[128x64] per block, 16x64 per wave =========
// 8 waves/block share a TDM-staged, double-buffered 32(K)x64(N) weight tile.
#define GEMM_PROLOG(APTR, WPTR)                                              \
  __shared__ u16 wtile[2][64 * 32];                                          \
  const int lane = threadIdx.x & 31;                                         \
  const int wave = threadIdx.x >> 5;                                         \
  const int g  = lane >> 4;                                                  \
  const int mr = lane & 15;                                                  \
  const int m0 = blockIdx.x * 128 + wave * 16;                               \
  const int n0 = blockIdx.y * 64;                                            \
  const unsigned lds0 = (unsigned)(size_t)(&wtile[0][0]);                    \
  const unsigned lds1 = (unsigned)(size_t)(&wtile[1][0]);                    \
  const u16* arow = (APTR) + (size_t)(m0 + mr) * D_;                         \
  const unsigned long long wga =                                             \
      (unsigned long long)(size_t)((WPTR) + (size_t)n0 * D_);                \
  const v8i g1 = tdm_desc_g1();                                              \
  v8f acc0 = {}, acc1 = {}, acc2 = {}, acc3 = {};                            \
  if (threadIdx.x < 32) tdm_issue(wga, lds0, g1);                            \
  const int NK = D_ / 32;                                                    \
  for (int kt = 0; kt < NK; ++kt) {                                          \
    const int k0 = kt * 32;                                                  \
    if (threadIdx.x < 32) {                                                  \
      if (kt + 1 < NK) {                                                     \
        tdm_issue(wga + (size_t)(kt + 1) * 64, ((kt + 1) & 1) ? lds1 : lds0, g1); \
        __builtin_amdgcn_s_wait_tensorcnt(1);                                \
      } else {                                                               \
        __builtin_amdgcn_s_wait_tensorcnt(0);                                \
      }                                                                      \
    }                                                                        \
    __syncthreads();                                                         \
    const u16* wt = wtile[kt & 1];                                           \
    Frag a;                                                                  \
    a.u[0] = *(const v4u*)(arow + k0 + g * 16);                              \
    a.u[1] = *(const v4u*)(arow + k0 + g * 16 + 8);                          \
    Frag b;                                                                  \
    b.u[0] = *(const v4u*)(wt + (0 * 16 + mr) * 32 + g * 16);                \
    b.u[1] = *(const v4u*)(wt + (0 * 16 + mr) * 32 + g * 16 + 8);            \
    acc0 = wmma_bf16(a, b, acc0);                                            \
    b.u[0] = *(const v4u*)(wt + (1 * 16 + mr) * 32 + g * 16);                \
    b.u[1] = *(const v4u*)(wt + (1 * 16 + mr) * 32 + g * 16 + 8);            \
    acc1 = wmma_bf16(a, b, acc1);                                            \
    b.u[0] = *(const v4u*)(wt + (2 * 16 + mr) * 32 + g * 16);                \
    b.u[1] = *(const v4u*)(wt + (2 * 16 + mr) * 32 + g * 16 + 8);            \
    acc2 = wmma_bf16(a, b, acc2);                                            \
    b.u[0] = *(const v4u*)(wt + (3 * 16 + mr) * 32 + g * 16);                \
    b.u[1] = *(const v4u*)(wt + (3 * 16 + mr) * 32 + g * 16 + 8);            \
    acc3 = wmma_bf16(a, b, acc3);                                            \
    __syncthreads();                                                         \
  }

// ---------------- stage 1: fused QKV projection ----------------
__global__ void qkv_proj(const u16* __restrict__ Xb,
                         const u16* __restrict__ Wqb,
                         const u16* __restrict__ Wkb,
                         const u16* __restrict__ Wvb,
                         u16* __restrict__ Q,    // (B,H,S,HD)
                         u16* __restrict__ Kt,   // (B,H,S,HD)
                         u16* __restrict__ Vt)   // (B,H,HD,S)  transposed
{
  const int z = blockIdx.z;
  const u16* W = (z == 0) ? Wqb : (z == 1) ? Wkb : Wvb;
  GEMM_PROLOG(Xb, W)

  for (int j = 0; j < 4; ++j) {
    const v8f c = (j == 0) ? acc0 : (j == 1) ? acc1 : (j == 2) ? acc2 : acc3;
    const int col = n0 + j * 16 + mr;
    const int h  = col >> 6;
    const int hd = col & 63;
    for (int r = 0; r < 8; ++r) {
      const int mg = m0 + g * 8 + r;
      const int bb = mg >> 11;
      const int s  = mg & (S_ - 1);
      const u16 v = f2bf(c[r]);
      if (z == 0)      Q [(((size_t)(bb * H_ + h) * S_) + s) * HD_ + hd] = v;
      else if (z == 1) Kt[(((size_t)(bb * H_ + h) * S_) + s) * HD_ + hd] = v;
      else             Vt[(((size_t)(bb * H_ + h) * HD_) + hd) * S_ + s] = v;
    }
  }
}

// ---------------- stage 3: output projection ----------------
__global__ void out_proj(const u16* __restrict__ Ab, const u16* __restrict__ Wob,
                         float* __restrict__ out)
{
  GEMM_PROLOG(Ab, Wob)

  for (int j = 0; j < 4; ++j) {
    const v8f c = (j == 0) ? acc0 : (j == 1) ? acc1 : (j == 2) ? acc2 : acc3;
    const int col = n0 + j * 16 + mr;
    for (int r = 0; r < 8; ++r) {
      const int mg = m0 + g * 8 + r;
      out[(size_t)mg * D_ + col] = c[r];
    }
  }
}

// ---------------- stage 2: causal flash attention ----------------
__global__ void attn_fwd(const u16* __restrict__ Q,
                         const u16* __restrict__ Kt,
                         const u16* __restrict__ Vt,
                         u16* __restrict__ attended /* (B,S,D) bf16 */)
{
  __shared__ u16 lds[4 * 16 * 32];               // 1KB P-staging per wave
  const int lane = threadIdx.x & 31;
  const int wave = threadIdx.x >> 5;
  const int g  = lane >> 4;
  const int mr = lane & 15;
  u16* pbuf = lds + wave * (16 * 32);

  const int wg = blockIdx.x * 4 + wave;          // 0 .. 4095
  const int bh = wg >> 7;                        // / (S_/16)
  const int q0 = (wg & 127) * 16;

  const u16* Qp = Q  + (size_t)bh * S_ * HD_;
  const u16* Kp = Kt + (size_t)bh * S_ * HD_;
  const u16* Vp = Vt + (size_t)bh * HD_ * S_;

  Frag qa0, qa1;
  {
    const u16* qrow = Qp + (size_t)(q0 + mr) * HD_;
    qa0.u[0] = *(const v4u*)(qrow + g * 16);
    qa0.u[1] = *(const v4u*)(qrow + g * 16 + 8);
    qa1.u[0] = *(const v4u*)(qrow + 32 + g * 16);
    qa1.u[1] = *(const v4u*)(qrow + 32 + g * 16 + 8);
  }

  float mrow[8], lrow[8], alp[8];
  v8f acc0 = {}, acc1 = {}, acc2 = {}, acc3 = {};
  for (int r = 0; r < 8; ++r) { mrow[r] = -__builtin_inff(); lrow[r] = 0.f; }

  const int ktiles = (q0 + 47) >> 5;             // cover keys 0 .. q0+15
  for (int kt = 0; kt < ktiles; ++kt) {
    const int kb = kt * 32;

    v8f c0 = {}, c1 = {};
    {
      Frag kbf;
      const u16* krow0 = Kp + (size_t)(kb + mr) * HD_;
      kbf.u[0] = *(const v4u*)(krow0 + g * 16);
      kbf.u[1] = *(const v4u*)(krow0 + g * 16 + 8);
      c0 = wmma_bf16(qa0, kbf, c0);
      kbf.u[0] = *(const v4u*)(krow0 + 32 + g * 16);
      kbf.u[1] = *(const v4u*)(krow0 + 32 + g * 16 + 8);
      c0 = wmma_bf16(qa1, kbf, c0);
      const u16* krow1 = Kp + (size_t)(kb + 16 + mr) * HD_;
      kbf.u[0] = *(const v4u*)(krow1 + g * 16);
      kbf.u[1] = *(const v4u*)(krow1 + g * 16 + 8);
      c1 = wmma_bf16(qa0, kbf, c1);
      kbf.u[0] = *(const v4u*)(krow1 + 32 + g * 16);
      kbf.u[1] = *(const v4u*)(krow1 + 32 + g * 16 + 8);
      c1 = wmma_bf16(qa1, kbf, c1);
    }

    for (int r = 0; r < 8; ++r) {
      const int mg = q0 + g * 8 + r;
      float s0 = c0[r] * 0.125f;                 // 1/sqrt(64)
      float s1 = c1[r] * 0.125f;
      if (kb + mr > mg)      s0 = -__builtin_inff();
      if (kb + 16 + mr > mg) s1 = -__builtin_inff();
      float rmax = fmaxf(s0, s1);
      for (int off = 8; off; off >>= 1) rmax = fmaxf(rmax, __shfl_xor(rmax, off, 32));
      const float mnew = fmaxf(mrow[r], rmax);
      const float a  = __expf(mrow[r] - mnew);
      const float p0 = __expf(s0 - mnew);
      const float p1 = __expf(s1 - mnew);
      float rs = p0 + p1;
      for (int off = 8; off; off >>= 1) rs += __shfl_xor(rs, off, 32);
      lrow[r] = lrow[r] * a + rs;
      mrow[r] = mnew;
      alp[r]  = a;
      pbuf[(g * 8 + r) * 32 + mr]      = f2bf(p0);
      pbuf[(g * 8 + r) * 32 + 16 + mr] = f2bf(p1);
    }

    Frag pa;
    pa.u[0] = *(const v4u*)(pbuf + mr * 32 + g * 16);
    pa.u[1] = *(const v4u*)(pbuf + mr * 32 + g * 16 + 8);

    for (int r = 0; r < 8; ++r) {
      acc0[r] *= alp[r]; acc1[r] *= alp[r]; acc2[r] *= alp[r]; acc3[r] *= alp[r];
    }
    Frag vb;
    const u16* vr;
    vr = Vp + (size_t)(0 * 16 + mr) * S_ + kb;
    vb.u[0] = *(const v4u*)(vr + g * 16); vb.u[1] = *(const v4u*)(vr + g * 16 + 8);
    acc0 = wmma_bf16(pa, vb, acc0);
    vr = Vp + (size_t)(1 * 16 + mr) * S_ + kb;
    vb.u[0] = *(const v4u*)(vr + g * 16); vb.u[1] = *(const v4u*)(vr + g * 16 + 8);
    acc1 = wmma_bf16(pa, vb, acc1);
    vr = Vp + (size_t)(2 * 16 + mr) * S_ + kb;
    vb.u[0] = *(const v4u*)(vr + g * 16); vb.u[1] = *(const v4u*)(vr + g * 16 + 8);
    acc2 = wmma_bf16(pa, vb, acc2);
    vr = Vp + (size_t)(3 * 16 + mr) * S_ + kb;
    vb.u[0] = *(const v4u*)(vr + g * 16); vb.u[1] = *(const v4u*)(vr + g * 16 + 8);
    acc3 = wmma_bf16(pa, vb, acc3);
  }

  const int bb = bh >> 4, h = bh & 15;
  for (int r = 0; r < 8; ++r) {
    const int s = q0 + g * 8 + r;
    const float inv = 1.0f / lrow[r];
    const size_t base = ((size_t)(bb * S_ + s)) * D_ + h * HD_ + mr;
    attended[base +  0] = f2bf(acc0[r] * inv);
    attended[base + 16] = f2bf(acc1[r] * inv);
    attended[base + 32] = f2bf(acc2[r] * inv);
    attended[base + 48] = f2bf(acc3[r] * inv);
  }
}

// ---------------- launch ----------------
extern "C" void kernel_launch(void* const* d_in, const int* in_sizes, int n_in,
                              void* d_out, int out_size, void* d_ws, size_t ws_size,
                              hipStream_t stream) {
  const float* x  = (const float*)d_in[0];
  const float* Wq = (const float*)d_in[1];
  const float* Wk = (const float*)d_in[2];
  const float* Wv = (const float*)d_in[3];
  const float* Wo = (const float*)d_in[4];

  char* ws = (char*)d_ws;
  size_t off = 0;
  u16* Xb  = (u16*)(ws + off); off += (size_t)M_ * D_ * 2;   // 8 MB (reused as attended)
  u16* Wqb = (u16*)(ws + off); off += (size_t)D_ * D_ * 2;   // 2 MB
  u16* Wkb = (u16*)(ws + off); off += (size_t)D_ * D_ * 2;
  u16* Wvb = (u16*)(ws + off); off += (size_t)D_ * D_ * 2;
  u16* Wob = (u16*)(ws + off); off += (size_t)D_ * D_ * 2;
  u16* Qb  = (u16*)(ws + off); off += (size_t)M_ * D_ * 2;   // 8 MB each
  u16* Ktb = (u16*)(ws + off); off += (size_t)M_ * D_ * 2;
  u16* Vtb = (u16*)(ws + off); off += (size_t)M_ * D_ * 2;
  u16* attended = Xb;   // Xb is dead after qkv_proj

  cvt_f32_bf16<<<8192, 256, 0, stream>>>(x,  Xb,  M_ * D_);
  cvt_f32_bf16<<<2048, 256, 0, stream>>>(Wq, Wqb, D_ * D_);
  cvt_f32_bf16<<<2048, 256, 0, stream>>>(Wk, Wkb, D_ * D_);
  cvt_f32_bf16<<<2048, 256, 0, stream>>>(Wv, Wvb, D_ * D_);
  cvt_f32_bf16<<<2048, 256, 0, stream>>>(Wo, Wob, D_ * D_);

  qkv_proj<<<dim3(M_ / 128, D_ / 64, 3), 256, 0, stream>>>(Xb, Wqb, Wkb, Wvb, Qb, Ktb, Vtb);

  attn_fwd<<<dim3((2 * H_ * (S_ / 16)) / 4), 128, 0, stream>>>(Qb, Ktb, Vtb, attended);

  out_proj<<<dim3(M_ / 128, D_ / 64), 256, 0, stream>>>(attended, Wob, (float*)d_out);
}
